// DisentangledRelationalCrossAttention_61993557950540
// MI455X (gfx1250) — compile-verified, matrix operand order checked by reading, not run on hardware
//
#include <hip/hip_runtime.h>
#include <hip/hip_bf16.h>
#include <math.h>

typedef __attribute__((ext_vector_type(16))) _Float16 v16h;
typedef __attribute__((ext_vector_type(8)))  _Float16 v8h;
typedef __attribute__((ext_vector_type(4)))  _Float16 v4h;
typedef __attribute__((ext_vector_type(8)))  float    v8f;

#define D_MODEL 512
#define N_HEADS 8
#define N_KV    4
#define HD      64
#define SEQ     512
#define BSZ     4

// ---------------------------------------------------------------------------
// CDNA5 async global->LDS copy (raw 16B per lane, tracked by ASYNCcnt).
// VDST = 32-bit LDS byte offset (low 32 bits of the generic LDS pointer,
// per ISA aperture layout), VADDR = 64-bit global address.
// ---------------------------------------------------------------------------
__device__ inline void async_g2l_b128(void* lds, const void* g) {
  unsigned off = (unsigned)(unsigned long)lds;
  asm volatile("global_load_async_to_lds_b128 %0, %1, off"
               :
               : "v"(off), "v"(g)
               : "memory");
}
__device__ inline void wait_async0() {
  asm volatile("s_wait_asynccnt 0" ::: "memory");
}

// ---------------------------------------------------------------------------
// WMMA fragment helpers (CDNA5 v_wmma_f32_16x16x32_f16, wave32)
// A (16x32, MxK): lane -> M = lane&15 ; VGPR v holds K pair
//   k0 = (v>=4?16:0) + (lane>>4)*8 + (v&3)*2        (ISA 7.12.2)
// B tiles are staged in LDS as [n][k] so the same gather pattern applies
// with N = lane&15.  Each half-fragment is 16 contiguous bytes -> the
// compiler lowers this to two ds_load_b128 per fragment.
// C/D (16x16 f32): lane -> N = lane&15 ; VGPR v -> M = v + 8*(lane>>4)
// ---------------------------------------------------------------------------
__device__ inline v16h frag_ld(const _Float16* p, int ld) {
  int lane = threadIdx.x & 31;
  int m = lane & 15, hi = lane >> 4;
  v16h a;
#pragma unroll
  for (int v = 0; v < 8; ++v) {
    int k0 = ((v & 4) << 2) + hi * 8 + ((v & 3) << 1);
    a[2 * v]     = p[m * ld + k0];
    a[2 * v + 1] = p[m * ld + k0 + 1];
  }
  return a;
}

// ---------------------------------------------------------------------------
// Generic tiled WMMA GEMM:  C[M,N] = scale * A[M,K] * B  (f16 in, f32 acc)
//  TRANSB=0 : B is [K,N] row-major (ldb = N-stride)
//  TRANSB=1 : B source is [N,K] row-major (ldb = K-stride)  -> Q*K^T case
// Batched via blockIdx.z: b = z/nH, h = z%nH (GQA kv head = h/rep).
// Block = 256 threads (8 waves), tile 64(M) x 64(N), K-chunks of 32.
// Wave w: m-tile = w&3, n-half = w>>2 (two 16x16 accumulators).
// A tile (and B tile when TRANSB) staged with async global->LDS b128.
// ---------------------------------------------------------------------------
template <bool TRANSB, bool OUTF16>
__global__ __launch_bounds__(256) void gemm_wmma(
    const _Float16* __restrict__ A, const _Float16* __restrict__ B,
    void* __restrict__ Cv, int K, int lda, int ldb, int ldc, float scale,
    int nH, int rep, long sAb, long sAh, long sBb, long sBh, long sCz) {
  __shared__ __align__(16) _Float16 As[64 * 32];
  __shared__ __align__(16) _Float16 Bs[64 * 32];  // stored [n][k]

  int z = blockIdx.z;
  int b = z / nH, h = z - b * nH;
  const _Float16* Ab = A + b * sAb + h * sAh;
  const _Float16* Bb = B + b * sBb + (long)(h / rep) * sBh;
  long coff = (long)z * sCz;

  int m0 = blockIdx.y * 64;
  int n0 = blockIdx.x * 64;
  int t = threadIdx.x;
  int w = t >> 5;
  int mt = w & 3;   // m-tile 0..3
  int nh = w >> 2;  // n-half 0..1

  v8f acc0 = {};
  v8f acc1 = {};

  for (int k0 = 0; k0 < K; k0 += 32) {
    {  // stage A tile [64][32] : one async 128-bit copy per thread
      int r = t >> 2, c = (t & 3) * 8;
      async_g2l_b128(&As[r * 32 + c], Ab + (long)(m0 + r) * lda + k0 + c);
    }
    if (TRANSB) {  // B source already [n][k] : async contiguous copy
      int r = t >> 2, c = (t & 3) * 8;
      async_g2l_b128(&Bs[r * 32 + c], Bb + (long)(n0 + r) * ldb + k0 + c);
    } else {  // B [k][n] : transpose while staging (VALU path)
      int r = t >> 3, c = (t & 7) * 8;  // r = k row, c = n col
      v8h tmp = *(const v8h*)(Bb + (long)(k0 + r) * ldb + n0 + c);
#pragma unroll
      for (int u = 0; u < 8; ++u) Bs[(c + u) * 32 + r] = tmp[u];
    }
    wait_async0();
    __syncthreads();

    v16h a  = frag_ld(&As[mt * 16 * 32], 32);
    v16h b0 = frag_ld(&Bs[(nh * 32) * 32], 32);
    v16h b1 = frag_ld(&Bs[(nh * 32 + 16) * 32], 32);
    acc0 = __builtin_amdgcn_wmma_f32_16x16x32_f16(false, a, false, b0,
                                                  (short)0, acc0, false, false);
    acc1 = __builtin_amdgcn_wmma_f32_16x16x32_f16(false, a, false, b1,
                                                  (short)0, acc1, false, false);
    __syncthreads();
  }

  int lane = t & 31, nl = lane & 15, hi = lane >> 4;
  int rowBase = m0 + mt * 16 + 8 * hi;
  int colBase = n0 + nh * 32 + nl;
#pragma unroll
  for (int v = 0; v < 8; ++v) {
    long idx = coff + (long)(rowBase + v) * ldc + colBase;
    if (OUTF16) {
      ((_Float16*)Cv)[idx]      = (_Float16)(acc0[v] * scale);
      ((_Float16*)Cv)[idx + 16] = (_Float16)(acc1[v] * scale);
    } else {
      ((float*)Cv)[idx]      = acc0[v] * scale;
      ((float*)Cv)[idx + 16] = acc1[v] * scale;
    }
  }
}

// ---------------------------------------------------------------------------
// Elementwise f32 -> f16 convert, and transpose-convert (for wv^T).
// ---------------------------------------------------------------------------
__global__ void cvt_f16(const float* __restrict__ in, _Float16* __restrict__ out,
                        int n) {
  int i = blockIdx.x * 256 + threadIdx.x;
  if (i < n) out[i] = (_Float16)in[i];
}

__global__ void transpose_cvt(const float* __restrict__ in,
                              _Float16* __restrict__ out, int rows, int cols) {
  int i = blockIdx.x * 256 + threadIdx.x;
  if (i < rows * cols) {
    int r = i / cols, c = i - r * cols;
    out[c * rows + r] = (_Float16)in[i];
  }
}

// ---------------------------------------------------------------------------
// Causal softmax over each 512-element row of attn (in place).
// Beyond-diagonal entries end up exactly 0 (matches softmax(-inf) * mask).
// ---------------------------------------------------------------------------
__global__ __launch_bounds__(256) void softmax_rows(float* __restrict__ attn,
                                                    const int* __restrict__ cf) {
  long row = blockIdx.x;
  int i = (int)(row & (SEQ - 1));
  float* p = attn + row * SEQ;
  int lim = (*cf) ? (i + 1) : SEQ;
  int t = threadIdx.x;

  float v0 = (2 * t < lim) ? p[2 * t] : -INFINITY;
  float v1 = (2 * t + 1 < lim) ? p[2 * t + 1] : -INFINITY;

  __shared__ float red[8];
  float m = fmaxf(v0, v1);
#pragma unroll
  for (int o = 16; o; o >>= 1) m = fmaxf(m, __shfl_xor(m, o));
  if ((t & 31) == 0) red[t >> 5] = m;
  __syncthreads();
  float mm = red[0];
#pragma unroll
  for (int u = 1; u < 8; ++u) mm = fmaxf(mm, red[u]);

  float e0 = (2 * t < lim) ? __expf(v0 - mm) : 0.f;
  float e1 = (2 * t + 1 < lim) ? __expf(v1 - mm) : 0.f;
  float s = e0 + e1;
#pragma unroll
  for (int o = 16; o; o >>= 1) s += __shfl_xor(s, o);
  __syncthreads();
  if ((t & 31) == 0) red[t >> 5] = s;
  __syncthreads();
  float ss = 0.f;
#pragma unroll
  for (int u = 0; u < 8; ++u) ss += red[u];
  float inv = 1.f / ss;
  p[2 * t]     = e0 * inv;
  p[2 * t + 1] = e1 * inv;
}

// ---------------------------------------------------------------------------
// Fused relational-value kernel, one block per (kv head g, query row i):
//   SV(j,d)  = symbols[i,j,:] @ wv[:, g*64 + d]          (on-the-fly, f16 WMMA)
//   out1[b,i,h,d] += sum_j attn[b,h,i,j]*rel[b,h,i,j] * SV(j,d),  h in {2g,2g+1}
// symbols streamed once from HBM (4 g-blocks for the same i share L2).
// LDS: wv g-slice [64][512] (64KB, staged via async b128) + symbols chunk
//      [32][512] (32KB, f32->f16 on VALU) + SV chunk as B-layout [64][32]
//      (4KB) + rca A-tile [16][32] (1KB).
// ---------------------------------------------------------------------------
__global__ __launch_bounds__(256) void relval_fused(
    const float* __restrict__ symbols, const _Float16* __restrict__ wvT,
    const float* __restrict__ attn, const float* __restrict__ rel,
    _Float16* __restrict__ out1) {
  __shared__ __align__(16) _Float16 wvg[64 * 512];
  __shared__ __align__(16) _Float16 symT[32 * 512];
  __shared__ __align__(16) _Float16 svT[64 * 32];
  __shared__ __align__(16) _Float16 rcaA[16 * 32];

  int g = blockIdx.x;  // kv head
  int i = blockIdx.y;  // query row
  int t = threadIdx.x;
  int w = t >> 5, lane = t & 31;
  int jm = w & 1;   // j sub-tile for SV compute
  int nt = w >> 1;  // n (=d) tile 0..3 for SV compute

  {  // stage wv^T g-slice once: [64 n][512 k], async global->LDS
    int n = t >> 2, c0 = (t & 3) * 128;
    const _Float16* src = wvT + (long)(g * 64 + n) * 512 + c0;
#pragma unroll
    for (int u = 0; u < 16; ++u)
      async_g2l_b128(&wvg[n * 512 + c0 + u * 8], src + u * 8);
  }
  wait_async0();

  v8f outAcc = {};

  for (int jc = 0; jc < 16; ++jc) {
    __syncthreads();  // previous chunk fully consumed / wvg staged

    {  // stage symbols[i, jc*32+jj, :] -> f16 LDS  [32][512]
      int jj = t >> 3, c0 = (t & 7) * 64;
      const float* src =
          symbols + ((long)i * SEQ + (jc * 32 + jj)) * D_MODEL + c0;
      if (jc < 15) __builtin_prefetch(src + 32 * D_MODEL, 0, 1);
#pragma unroll
      for (int u = 0; u < 16; ++u) {
        float4 f = *(const float4*)(src + u * 4);
        v4h hv = {(_Float16)f.x, (_Float16)f.y, (_Float16)f.z, (_Float16)f.w};
        *(v4h*)&symT[jj * 512 + c0 + u * 4] = hv;
      }
    }
    {  // stage rca A-tile: rows r = (b, h-in-group), cols j  (rows 8..15 = 0)
      int r = t >> 4, j2 = (t & 15) * 2;
      _Float16 h0 = (_Float16)0.f, h1 = (_Float16)0.f;
      if (r < 8) {
        int b = r >> 1, h = g * 2 + (r & 1);
        long base = (((long)(b * N_HEADS + h) * SEQ) + i) * SEQ + jc * 32 + j2;
        h0 = (_Float16)(attn[base] * rel[base]);
        h1 = (_Float16)(attn[base + 1] * rel[base + 1]);
      }
      rcaA[r * 32 + j2]     = h0;
      rcaA[r * 32 + j2 + 1] = h1;
    }
    __syncthreads();

    // SV chunk: wave (jm,nt) computes 16x16 tile over K=512
    v8f sv = {};
#pragma unroll
    for (int k = 0; k < 16; ++k) {
      v16h a  = frag_ld(&symT[jm * 16 * 512 + k * 32], 512);
      v16h bb = frag_ld(&wvg[nt * 16 * 512 + k * 32], 512);
      sv = __builtin_amdgcn_wmma_f32_16x16x32_f16(false, a, false, bb, (short)0,
                                                  sv, false, false);
    }
    {  // write SV tile to LDS in B layout [n][j]
      int nl = lane & 15, hi = lane >> 4;
#pragma unroll
      for (int v = 0; v < 8; ++v)
        svT[(nt * 16 + nl) * 32 + jm * 16 + 8 * hi + v] = (_Float16)sv[v];
    }
    __syncthreads();

    if (w < 4) {  // outer accumulate: out16[16, nt16] += rca[16,32j] @ SV[32j,16]
      v16h a  = frag_ld(rcaA, 32);
      v16h bb = frag_ld(&svT[w * 16 * 32], 32);
      outAcc = __builtin_amdgcn_wmma_f32_16x16x32_f16(false, a, false, bb,
                                                      (short)0, outAcc, false,
                                                      false);
    }
  }
  __syncthreads();

  // store out1[b,i,h, w*16 + nl] from rows r = v (valid r<8 live in hi==0)
  if (w < 4 && (lane >> 4) == 0) {
    int nl = lane & 15;
#pragma unroll
    for (int v = 0; v < 8; ++v) {
      int b = v >> 1, h = g * 2 + (v & 1);
      long off = (((long)(b * SEQ) + i) * N_HEADS + h) * (long)HD + w * 16 + nl;
      out1[off] = (_Float16)outAcc[v];
    }
  }
}

// ---------------------------------------------------------------------------
// Launch
// ---------------------------------------------------------------------------
extern "C" void kernel_launch(void* const* d_in, const int* in_sizes, int n_in,
                              void* d_out, int out_size, void* d_ws,
                              size_t ws_size, hipStream_t stream) {
  (void)in_sizes; (void)n_in; (void)out_size; (void)ws_size;
  const float* x       = (const float*)d_in[0];
  const float* symbols = (const float*)d_in[1];
  const float* wq_a    = (const float*)d_in[2];
  const float* wk_a    = (const float*)d_in[3];
  const float* wq_r    = (const float*)d_in[4];
  const float* wk_r    = (const float*)d_in[5];
  const float* wv      = (const float*)d_in[6];
  const float* wo      = (const float*)d_in[7];
  const int*   causal  = (const int*)d_in[8];

  float* out  = (float*)d_out;                       // [4,512,512]
  float* attn = out + (long)BSZ * SEQ * D_MODEL;     // [4,8,512,512]
  float* rel  = attn + (long)BSZ * N_HEADS * SEQ * SEQ;

  _Float16* p    = (_Float16*)d_ws;
  _Float16* xh   = p; p += 2048 * 512;
  _Float16* wqah = p; p += 512 * 512;
  _Float16* wkah = p; p += 512 * 256;
  _Float16* wqrh = p; p += 512 * 512;
  _Float16* wkrh = p; p += 512 * 256;
  _Float16* wvT  = p; p += 256 * 512;   // wv transposed: [n=256][k=512]
  _Float16* woh  = p; p += 512 * 512;
  _Float16* qah  = p; p += 2048 * 512;
  _Float16* kah  = p; p += 2048 * 256;
  _Float16* qrh  = p; p += 2048 * 512;
  _Float16* krh  = p; p += 2048 * 256;
  _Float16* out1 = p; p += 2048 * 512;

  dim3 blk(256);

  // f32 -> f16 conversions
  cvt_f16<<<4096, blk, 0, stream>>>(x, xh, 2048 * 512);
  cvt_f16<<<1024, blk, 0, stream>>>(wq_a, wqah, 512 * 512);
  cvt_f16<<<512,  blk, 0, stream>>>(wk_a, wkah, 512 * 256);
  cvt_f16<<<1024, blk, 0, stream>>>(wq_r, wqrh, 512 * 512);
  cvt_f16<<<512,  blk, 0, stream>>>(wk_r, wkrh, 512 * 256);
  cvt_f16<<<1024, blk, 0, stream>>>(wo, woh, 512 * 512);
  transpose_cvt<<<512, blk, 0, stream>>>(wv, wvT, 512, 256);

  // projections: [2048,512] @ [512,N] -> f16
  gemm_wmma<false, true><<<dim3(8, 32, 1), blk, 0, stream>>>(
      xh, wqah, qah, 512, 512, 512, 512, 1.f, 1, 1, 0, 0, 0, 0, 0);
  gemm_wmma<false, true><<<dim3(4, 32, 1), blk, 0, stream>>>(
      xh, wkah, kah, 512, 512, 256, 256, 1.f, 1, 1, 0, 0, 0, 0, 0);
  gemm_wmma<false, true><<<dim3(8, 32, 1), blk, 0, stream>>>(
      xh, wqrh, qrh, 512, 512, 512, 512, 1.f, 1, 1, 0, 0, 0, 0, 0);
  gemm_wmma<false, true><<<dim3(4, 32, 1), blk, 0, stream>>>(
      xh, wkrh, krh, 512, 512, 256, 256, 1.f, 1, 1, 0, 0, 0, 0, 0);

  // batched scores: per (b,h)  Q[512,64] @ K^T -> f32 into d_out regions
  const float scale = 0.125f;  // 1/sqrt(64)
  gemm_wmma<true, false><<<dim3(8, 8, 32), blk, 0, stream>>>(
      qah, kah, attn, 64, 512, 256, 512, scale, N_HEADS, 2,
      262144, 64, 131072, 64, 262144);
  gemm_wmma<true, false><<<dim3(8, 8, 32), blk, 0, stream>>>(
      qrh, krh, rel, 64, 512, 256, 512, scale, N_HEADS, 2,
      262144, 64, 131072, 64, 262144);

  // causal softmax on attn (in place, reads device is_causal scalar)
  softmax_rows<<<BSZ * N_HEADS * SEQ, blk, 0, stream>>>(attn, causal);

  // fused sv-compute + relational value gather -> out1 (f16)
  relval_fused<<<dim3(N_KV, SEQ, 1), blk, 0, stream>>>(symbols, wvT, attn, rel,
                                                       out1);

  // final projection: out1 @ wo -> f32 d_out[0 : 4*512*512]
  gemm_wmma<false, false><<<dim3(8, 32, 1), blk, 0, stream>>>(
      out1, woh, out, 512, 512, 512, 512, 1.f, 1, 1, 0, 0, 0, 0, 0);
}
